// Attention_layer_44676249813094
// MI455X (gfx1250) — matrix operand order repaired and checked by reading, not compiled, not run on hardware
//
#include <hip/hip_runtime.h>

typedef __attribute__((ext_vector_type(16))) _Float16 v16h;
typedef __attribute__((ext_vector_type(8)))  float    v8f;
typedef __attribute__((ext_vector_type(4)))  int      v4i;

#define N_ 64
#define L_ 576
#define T_ 32
#define D_ 2048

#define GLOBAL_AS __attribute__((address_space(1)))
#define LDS_AS    __attribute__((address_space(3)))

// ---- async global->LDS staging (gfx1250), guarded so fallback still compiles
#if defined(__has_builtin)
#  if __has_builtin(__builtin_amdgcn_global_load_async_to_lds_b128)
#    define HAVE_ASYNC_LDS 1
#  endif
#  if __has_builtin(__builtin_amdgcn_s_wait_asynccnt)
#    define WAIT_ASYNC(n) __builtin_amdgcn_s_wait_asynccnt(n)
#  endif
#endif
#ifndef HAVE_ASYNC_LDS
#  define HAVE_ASYNC_LDS 0
#endif
#ifndef WAIT_ASYNC
#  define WAIT_ASYNC(n) asm volatile("s_wait_asynccnt %0" ::"n"(n))
#endif

// ---------------------------------------------------------------------------
// Kernel 1: f1e = relu(feature_1) streamed out; s1[n,l] = dot(f1e[n,l,:], w).
// One block per (n,l) row; 256 threads x 8 floats = 2048 = D.
// ---------------------------------------------------------------------------
__global__ __launch_bounds__(256) void k_relu_dot(const float* __restrict__ f1,
                                                  const float* __restrict__ w,
                                                  float* __restrict__ out_f1e,
                                                  float* __restrict__ s1) {
    __shared__ float red[256];
    const int row = blockIdx.x;          // n*L + l
    const int tid = threadIdx.x;
    const float* src = f1      + (size_t)row * D_;
    float*       dst = out_f1e + (size_t)row * D_;

    float acc = 0.f;
#pragma unroll
    for (int j = 0; j < 2; ++j) {
        const int idx = (tid + j * 256) * 4;
        float4 v  = *(const float4*)(src + idx);
        float4 wv = *(const float4*)(w + idx);
        v.x = fmaxf(v.x, 0.f); v.y = fmaxf(v.y, 0.f);
        v.z = fmaxf(v.z, 0.f); v.w = fmaxf(v.w, 0.f);
        *(float4*)(dst + idx) = v;
        acc += v.x * wv.x + v.y * wv.y + v.z * wv.z + v.w * wv.w;
    }
    red[tid] = acc;
    __syncthreads();
#pragma unroll
    for (int s = 128; s > 0; s >>= 1) {
        if (tid < s) red[tid] += red[tid + s];
        __syncthreads();
    }
    if (tid == 0) s1[row] = red[0];
}

// ---------------------------------------------------------------------------
// Kernel 2: p[n,:] = softmax(s1[n,:]) over L.  (s2 and fc_b are constant in l
// and cancel inside the softmax, so att[n,t,:] == p[n,:] for every t.)
// Writes p to workspace and att = p replicated T times.
// ---------------------------------------------------------------------------
__global__ __launch_bounds__(1024) void k_softmax_att(const float* __restrict__ s1,
                                                      float* __restrict__ p,
                                                      float* __restrict__ out_att) {
    __shared__ float red[1024];
    __shared__ float m_s, z_s;
    const int n = blockIdx.x, tid = threadIdx.x;

    const float v = (tid < L_) ? s1[n * L_ + tid] : -3.4e38f;
    red[tid] = v;
    __syncthreads();
#pragma unroll
    for (int s = 512; s > 0; s >>= 1) {
        if (tid < s) red[tid] = fmaxf(red[tid], red[tid + s]);
        __syncthreads();
    }
    if (tid == 0) m_s = red[0];
    __syncthreads();

    const float e = (tid < L_) ? __expf(v - m_s) : 0.f;
    red[tid] = e;
    __syncthreads();
#pragma unroll
    for (int s = 512; s > 0; s >>= 1) {
        if (tid < s) red[tid] += red[tid + s];
        __syncthreads();
    }
    if (tid == 0) z_s = red[0];
    __syncthreads();

    if (tid < L_) {
        const float pv = e / z_s;
        p[n * L_ + tid] = pv;
        for (int t = 0; t < T_; ++t)
            out_att[((size_t)n * T_ + t) * L_ + tid] = pv;   // coalesced per t
    }
}

// ---------------------------------------------------------------------------
// Kernel 3: g[n,d] = sum_l p[n,l]*f1e[n,l,d] via V_WMMA_F32_16X16X32_F16
// (A = p broadcast to 16 rows, f16; B = f1e tile, f16; f32 accumulate),
// then f2_out = relu(relu(feature_2) + g).
// Block = (n, 128-wide d chunk); 8 waves x 16 columns each.
// LDS double-buffered; tiles staged with async global->LDS when available.
// ---------------------------------------------------------------------------
__global__ __launch_bounds__(256) void k_pv_out(const float* __restrict__ f1e,
                                                const float* __restrict__ f2,
                                                const float* __restrict__ p,
                                                float* __restrict__ out_f2) {
    __shared__ float p_lds[L_];
    __shared__ float tile[2][32][128 + 4];   // double-buffered, padded
    __shared__ float g_lds[128];

    const int tid  = threadIdx.x;
    const int lane = tid & 31;
    const int wave = tid >> 5;                 // 0..7
    const int n    = blockIdx.x >> 4;
    const int d0   = (blockIdx.x & 15) * 128;

    for (int i = tid; i < L_; i += 256) p_lds[i] = p[n * L_ + i];

    const float* src = f1e + (size_t)n * L_ * D_ + d0;

    // Stage one 32x128 f32 tile into tile[buf]: 1024 float4, 4 per thread.
    auto stage = [&](int buf, int k0) {
#pragma unroll
        for (int j = 0; j < 4; ++j) {
            const int f4i = tid + j * 256;
            const int r   = f4i >> 5;          // 32 float4 per row
            const int c4  = (f4i & 31) << 2;
            const float* g = src + (size_t)(k0 + r) * D_ + c4;
#if HAVE_ASYNC_LDS
            __builtin_amdgcn_global_load_async_to_lds_b128(
                (GLOBAL_AS v4i*)g,
                (LDS_AS v4i*)&tile[buf][r][c4], 0, 0);
#else
            float4 v = *(const float4*)g;
            tile[buf][r][c4 + 0] = v.x; tile[buf][r][c4 + 1] = v.y;
            tile[buf][r][c4 + 2] = v.z; tile[buf][r][c4 + 3] = v.w;
#endif
        }
    };

    constexpr int NSTEP = L_ / 32;             // 18 k-steps

    // Fragment index maps, contiguous k = 16*(lane/16) + e (16-bit A 16x32, B 32x16):
    const int kb_a = (lane < 16) ? 0 : 16;     // A: half-wave K base
    const int kb_b = (lane < 16) ? 0 : 16;     // B: half-wave K base
    const int col  = (wave << 4) + (lane & 15);

    stage(0, 0);                               // prologue fill

    v8f c = {};
    for (int step = 0; step < NSTEP; ++step) {
        const int cur = step & 1;
        const int k0  = step * 32;
        if (step + 1 < NSTEP) {
            stage(cur ^ 1, k0 + 32);           // overlap next tile with compute
#if HAVE_ASYNC_LDS
            WAIT_ASYNC(4);                     // in-order: current tile's 4 done
#endif
#if !HAVE_ASYNC_LDS
            __builtin_prefetch(src + (size_t)(k0 + 64 + (tid >> 5)) * D_ + ((tid & 31) << 2), 0, 1);
#endif
        } else {
#if HAVE_ASYNC_LDS
            WAIT_ASYNC(0);
#endif
        }
        __syncthreads();                       // all waves' tile[cur] visible

        // A fragment: 16 identical rows of p[k0..k0+32)
        v16h a, b;
#pragma unroll
        for (int e = 0; e < 8; ++e) {
            a[e]     = (_Float16)p_lds[k0 + kb_a + e];
            a[e + 8] = (_Float16)p_lds[k0 + 8 + kb_a + e];
        }
        // B fragment: column 'col', K = kb_b + e
#pragma unroll
        for (int e = 0; e < 16; ++e)
            b[e] = (_Float16)tile[cur][kb_b + e][col];

        c = __builtin_amdgcn_wmma_f32_16x16x32_f16(false, a, false, b, (short)0, c,
                                                   false, false);
        __syncthreads();                       // reads of tile[cur] done -> reusable
    }

    // All 16 D rows identical; lane<16, VGPR0 = D[M=0][N=lane] = g[col].
    if (lane < 16) g_lds[col] = c[0];
    __syncthreads();

    const float* f2n = f2     + (size_t)n * T_ * D_ + d0;
    float*       on  = out_f2 + (size_t)n * T_ * D_ + d0;
    for (int idx = tid; idx < T_ * 128; idx += 256) {
        const int t = idx >> 7, i = idx & 127;
        float v = f2n[(size_t)t * D_ + i];
        v = fmaxf(v, 0.f);                 // relu(feature_2)
        v = fmaxf(v + g_lds[i], 0.f);      // relu(f2e + f_hat)
        on[(size_t)t * D_ + i] = v;
    }
}

// ---------------------------------------------------------------------------
extern "C" void kernel_launch(void* const* d_in, const int* in_sizes, int n_in,
                              void* d_out, int out_size, void* d_ws, size_t ws_size,
                              hipStream_t stream) {
    const float* f1 = (const float*)d_in[0];   // (N,L,D)
    const float* f2 = (const float*)d_in[1];   // (N,T,D)
    const float* w  = (const float*)d_in[2];   // (D,)
    // d_in[3] = fc_b: constant along the softmax axis -> cancels; unused.

    float* out_f1e = (float*)d_out;                          // N*L*D
    float* out_f2  = out_f1e + (size_t)N_ * L_ * D_;         // N*T*D
    float* out_att = out_f2  + (size_t)N_ * T_ * D_;         // N*T*L

    float* s1 = (float*)d_ws;          // N*L
    float* pp = s1 + N_ * L_;          // N*L

    k_relu_dot   <<<N_ * L_, 256,  0, stream>>>(f1, w, out_f1e, s1);
    k_softmax_att<<<N_,      1024, 0, stream>>>(s1, pp, out_att);
    k_pv_out     <<<N_ * 16, 256,  0, stream>>>(out_f1e, f2, pp, out_f2);
}
